// BorderPostProcessor_21234318311585
// MI455X (gfx1250) — compile-verified, matrix operand order checked by reading, not run on hardware
//
#include <hip/hip_runtime.h>
#include <hip/hip_bf16.h>

// ---------------- problem constants (match reference) ----------------
#define NN 16
#define KK 80
#define HH 100
#define WWD 152
#define PP (HH*WWD)          // 15200
#define LL (PP*KK)           // 1,216,000 candidates / image
#define MM 1024              // padded working set (>= PRE_NMS_TOP_N)
#define TOPN 1000            // PRE_NMS_TOP_N
#define POSTN 100            // FPN_POST_NMS_TOP_N
#define PRE_THRESH 0.05f
#define NMS_THRESH 0.6f

typedef __attribute__((ext_vector_type(16))) _Float16 v16h;
typedef __attribute__((ext_vector_type(8)))  float    v8f;

__device__ __forceinline__ float sigmoidf_(float x) { return 1.0f / (1.0f + __expf(-x)); }

// order-preserving float->uint map (total order, -1.0 .. 1.0 values here)
__device__ __forceinline__ unsigned ordkey(float f) {
    unsigned u = __float_as_uint(f);
    return (u & 0x80000000u) ? ~u : (u | 0x80000000u);
}

// ---------------- K1: fused score computation (streaming, HBM-bound) ---------
// masked[n][p*K+k] = (s>0.05 ? s*sig(bcls) : -1),  s = sqrt(sig(cls)*sig(ctr))
__global__ void k1_fuse(const float* __restrict__ box_cls,
                        const float* __restrict__ box_center,
                        const float* __restrict__ border_cls,
                        float* __restrict__ masked) {
    long i = (long)blockIdx.x * blockDim.x + threadIdx.x;
    if (i >= (long)NN * KK * PP) return;
    int n = (int)(i / ((long)KK * PP));
    int r = (int)(i % ((long)KK * PP));
    int k = r / PP;
    int p = r % PP;
    float s = sqrtf(sigmoidf_(box_cls[i]) * sigmoidf_(box_center[(size_t)n * PP + p]));
    float prob = s * sigmoidf_(border_cls[i]);
    masked[(size_t)n * LL + (size_t)p * KK + k] = (s > PRE_THRESH) ? prob : -1.0f;
}

// ---------------- K2: deterministic top-1000 selection per image -------------
// binary-search value threshold, deterministic index tie-break, compact,
// canonicalize with in-LDS bitonic sort on unique (value,index) keys.
__global__ void k2_select(const float* __restrict__ masked,
                          int* __restrict__ topIdx, float* __restrict__ topVal) {
    const int n = blockIdx.x, tid = threadIdx.x;
    const float* img = masked + (size_t)n * LL;
    __shared__ unsigned long long sKey[MM];
    __shared__ unsigned s_cnt, s_lo, s_hi;
    __shared__ int s_pos;

    // --- T* = value of the 1000th-largest element (ordkey space) ---
    if (tid == 0) { s_lo = 0u; s_hi = 0xFFFFFFFFu; }
    __syncthreads();
    for (int it = 0; it < 33; ++it) {
        unsigned lo = s_lo, hi = s_hi;
        if (lo >= hi) break;
        unsigned mid = lo + ((hi - lo) >> 1) + 1u;
        __syncthreads();
        if (tid == 0) s_cnt = 0u;
        __syncthreads();
        unsigned c = 0;
        for (int i = tid; i < LL; i += 1024) c += (ordkey(img[i]) >= mid);
        if (c) atomicAdd(&s_cnt, c);
        __syncthreads();
        if (tid == 0) { if (s_cnt >= TOPN) s_lo = mid; else s_hi = mid - 1u; }
        __syncthreads();
    }
    const unsigned Tstar = s_lo;

    // --- G = count strictly greater; need = TOPN - G ties taken by lowest idx ---
    __syncthreads();
    if (tid == 0) s_cnt = 0u;
    __syncthreads();
    {
        unsigned c = 0;
        for (int i = tid; i < LL; i += 1024) c += (ordkey(img[i]) > Tstar);
        if (c) atomicAdd(&s_cnt, c);
    }
    __syncthreads();
    const unsigned G = s_cnt;
    const unsigned need = TOPN - G;

    // --- X* = smallest index bound s.t. #{==T*, idx<X*} >= need ---
    if (tid == 0) { s_lo = 0u; s_hi = LL; }
    __syncthreads();
    for (int it = 0; it < 33; ++it) {
        unsigned lo = s_lo, hi = s_hi;
        if (lo >= hi) break;
        unsigned mid = (lo + hi) >> 1;
        __syncthreads();
        if (tid == 0) s_cnt = 0u;
        __syncthreads();
        unsigned c = 0;
        for (int i = tid; i < LL; i += 1024)
            c += ((ordkey(img[i]) == Tstar) && ((unsigned)i < mid));
        if (c) atomicAdd(&s_cnt, c);
        __syncthreads();
        if (tid == 0) { if (s_cnt >= need) s_hi = mid; else s_lo = mid + 1u; }
        __syncthreads();
    }
    const unsigned Xstar = s_lo;

    // --- compact (atomic order irrelevant: sort canonicalizes) ---
    if (tid == 0) s_pos = 0;
    __syncthreads();
    for (int i = tid; i < LL; i += 1024) {
        unsigned u = ordkey(img[i]);
        if ((u > Tstar) || (u == Tstar && (unsigned)i < Xstar)) {
            int p = atomicAdd(&s_pos, 1);
            if (p < MM) sKey[p] = ((unsigned long long)(~u) << 32) | (unsigned)i;
        }
    }
    __syncthreads();
    if (tid >= s_pos) sKey[tid] = 0xFFFFFFFFFFFFFFFFull;   // sentinel sorts last

    // --- bitonic sort ascending: rank = ~ordkey => value desc, index asc ---
    for (int k2 = 2; k2 <= MM; k2 <<= 1)
        for (int j = k2 >> 1; j > 0; j >>= 1) {
            __syncthreads();
            int ixj = tid ^ j;
            if (ixj > tid) {
                unsigned long long va = sKey[tid], vb = sKey[ixj];
                bool asc = ((tid & k2) == 0);
                if ((va > vb) == asc) { sKey[tid] = vb; sKey[ixj] = va; }
            }
        }
    __syncthreads();

    unsigned long long kkey = sKey[tid];
    int flat = -1; float val = -1.0f;
    if ((unsigned)(kkey >> 32) != 0xFFFFFFFFu) {
        flat = (int)(kkey & 0xFFFFFFFFu);
        val = img[flat];
    }
    topIdx[n * MM + tid] = flat;
    topVal[n * MM + tid] = val;
}

// ---------------- K3: decode boxes, per-image max coord, class offsets -------
__global__ void k3_decode(const int* __restrict__ topIdx, const float* __restrict__ topVal,
                          const float* __restrict__ border_delta,
                          const float* __restrict__ bd_based,
                          float* __restrict__ decBox, float* __restrict__ offBox,
                          _Float16* __restrict__ areaH, float* __restrict__ score,
                          int* __restrict__ clsA, unsigned* __restrict__ validw,
                          unsigned* __restrict__ maskb) {
    const int n = blockIdx.x, m = threadIdx.x;
    __shared__ float sm[MM];
    int flat = topIdx[n * MM + m];
    float val = topVal[n * MM + m];
    float b0 = 0.f, b1 = 0.f, b2 = 0.f, b3 = 0.f; int c = 0;
    if (flat >= 0) {
        int p = flat / KK; c = flat % KK;
        const float* bb = bd_based + ((size_t)n * PP + p) * 4;
        float r0 = border_delta[((size_t)n * 4 + 0) * PP + p];
        float r1 = border_delta[((size_t)n * 4 + 1) * PP + p];
        float r2 = border_delta[((size_t)n * 4 + 2) * PP + p];
        float r3 = border_delta[((size_t)n * 4 + 3) * PP + p];
        float w = bb[2] - bb[0], h = bb[3] - bb[1];
        b0 = bb[0] + r0 * 0.1f * w;
        b1 = bb[1] + r1 * 0.1f * h;
        b2 = bb[2] + r2 * 0.2f * w;
        b3 = bb[3] + r3 * 0.2f * h;
    }
    bool valid = (m < TOPN) && (val > 0.0f);
    // bxs.max() over the 1000 real rows
    sm[m] = (m < TOPN) ? fmaxf(fmaxf(b0, b1), fmaxf(b2, b3)) : -3.4e38f;
    __syncthreads();
    for (int s = 512; s > 0; s >>= 1) {
        if (m < s) sm[m] = fmaxf(sm[m], sm[m + s]);
        __syncthreads();
    }
    float off = (float)c * (sm[0] + 1.0f);

    size_t b = ((size_t)n * MM + m) * 4;
    decBox[b + 0] = b0; decBox[b + 1] = b1; decBox[b + 2] = b2; decBox[b + 3] = b3;
    offBox[b + 0] = b0 + off; offBox[b + 1] = b1 + off;
    offBox[b + 2] = b2 + off; offBox[b + 3] = b3 + off;
    areaH[n * MM + m] = (_Float16)((b2 - b0) * (b3 - b1));   // fits FP16 (<=16K)
    score[n * MM + m] = valid ? sqrtf(val) : 0.0f;
    clsA[n * MM + m]  = c;

    unsigned long long bv = __ballot(valid ? 1 : 0);      // wave32: bits 0..31
    if ((m & 31) == 0) validw[n * 32 + (m >> 5)] = (unsigned)bv;
    for (int k = m; k < MM * 32; k += 1024) maskb[(size_t)n * MM * 32 + k] = 0u;
}

// ---------------- K4: IoU>thresh bitmask; WMMA computes area_i+area_j --------
// A(16x32 f16): K0=area_i, K1=1 ; B(32x16 f16): K0=1, K1=area_j
// D = A*B = area_i + area_j on the matrix pipe; VALU co-executes intersection.
// iou > t  <=>  inter > t*(areaSum - inter + eps)   (denominator positive)
__global__ void k4_iou(const _Float16* __restrict__ areaH,
                       const float* __restrict__ offBox,
                       unsigned* __restrict__ maskb) {
    const int n = blockIdx.x;
    const int wave = threadIdx.x >> 5, lane = threadIdx.x & 31;
    const int wg = blockIdx.y * 8 + wave;                 // 0..127 waves / image
    const _Float16* ar = areaH + (size_t)n * MM;
    const float4* ob4 = reinterpret_cast<const float4*>(offBox) + (size_t)n * MM;
    unsigned* mb = maskb + (size_t)n * MM * 32;

    const int mhi = (lane >> 4) << 3;          // 0 for lanes 0-15, 8 for 16-31
    const bool writer = ((lane & 15) == 0);    // lanes 0 and 16 emit row words
    const unsigned halfsel = (lane >> 4) << 4; // 0 or 16: which ballot half

    for (int s = 0; s < 32; ++s) {             // 32 tiles / wave, 4096 = 64x64
        int T  = wg * 32 + s;
        int i0 = (T >> 6) << 4, j0 = (T & 63) << 4;
        v16h A = {}; v16h B = {};
        if (lane < 16) {
            A[0] = ar[i0 + lane]; A[1] = (_Float16)1.0f;
            B[0] = (_Float16)1.0f; B[1] = ar[j0 + lane];
        }
        v8f C = {};
        C = __builtin_amdgcn_wmma_f32_16x16x32_f16(false, A, false, B,
                                                   (short)0, C, false, false);
        int jj = j0 + (lane & 15);
        float4 Bj = ob4[jj];
#pragma unroll
        for (int r = 0; r < 8; ++r) {
            int ii = i0 + mhi + r;
            float4 Bi = ob4[ii];
            float ltx = fmaxf(Bi.x, Bj.x), lty = fmaxf(Bi.y, Bj.y);
            float rbx = fminf(Bi.z, Bj.z), rby = fminf(Bi.w, Bj.w);
            float inter = fmaxf(rbx - ltx, 0.f) * fmaxf(rby - lty, 0.f);
            // division-free threshold test (denominator > 0)
            bool hit = inter > NMS_THRESH * (C[r] - inter + 1e-9f);
            unsigned long long bal = __ballot(hit ? 1 : 0);
            unsigned bits = (unsigned)((bal >> halfsel) & 0xFFFFull);
            if (writer)
                atomicOr(&mb[(size_t)ii * 32 + (j0 >> 5)], bits << (j0 & 16));
        }
    }
}

// ---------------- K5: sequential greedy NMS over bitmask rows ----------------
__global__ void k5_nms(const unsigned* __restrict__ validw,
                       const unsigned* __restrict__ maskb,
                       unsigned* __restrict__ supw) {
    const int n = blockIdx.x, t = threadIdx.x;            // block of 32 (1 wave)
    __shared__ unsigned sup[32];
    sup[t] = ~validw[n * 32 + t];                         // sup0 = !valid
    __syncthreads();
    const unsigned* mb = maskb + (size_t)n * MM * 32;
    for (int i = 0; i < TOPN; ++i) {
        int wi = i >> 5, bit = i & 31;
        unsigned si = sup[wi];
        __syncthreads();
        if (!((si >> bit) & 1u)) {
            unsigned mrow = mb[(size_t)i * 32 + t];
            if (t < wi) mrow = 0u;
            else if (t == wi) mrow &= ~((2u << bit) - 1u); // keep only j > i
            sup[t] |= mrow;
        }
        __syncthreads();
    }
    supw[n * 32 + t] = sup[t];
}

// ---------------- K6: post-NMS top-100 gate, clip, min-size, emit ------------
__global__ void k6_final(const float* __restrict__ decBox, const float* __restrict__ score,
                         const int* __restrict__ clsA, const unsigned* __restrict__ validw,
                         const unsigned* __restrict__ supw, const int* __restrict__ img_sz,
                         float* __restrict__ out) {
    const int n = blockIdx.x, m = threadIdx.x;
    __shared__ float ssc[MM];
    __shared__ unsigned s_cnt, s_lo, s_hi;
    bool valid = (validw[n * 32 + (m >> 5)] >> (m & 31)) & 1u;
    bool sup   = (supw[n * 32 + (m >> 5)] >> (m & 31)) & 1u;
    bool keep = valid && !sup;
    float sc_m = score[n * MM + m];
    float sc = keep ? sc_m : -1.0f;
    ssc[m] = sc;
    if (m == 0) s_cnt = 0u;
    __syncthreads();
    if (keep) atomicAdd(&s_cnt, 1u);
    __syncthreads();
    unsigned ndet = s_cnt;

    // kth-largest score (100th) via ordered-uint binary search
    if (m == 0) { s_lo = 0u; s_hi = 0xFFFFFFFFu; }
    __syncthreads();
    for (int it = 0; it < 33; ++it) {
        unsigned lo = s_lo, hi = s_hi;
        if (lo >= hi) break;
        unsigned mid = lo + ((hi - lo) >> 1) + 1u;
        __syncthreads();
        if (m == 0) s_cnt = 0u;
        __syncthreads();
        if (ordkey(ssc[m]) >= mid) atomicAdd(&s_cnt, 1u);
        __syncthreads();
        if (m == 0) { if (s_cnt >= POSTN) s_lo = mid; else s_hi = mid - 1u; }
        __syncthreads();
    }
    if (ndet > POSTN) keep = keep && (ordkey(sc) >= s_lo);

    float fh = (float)img_sz[n * 2 + 0] - 1.0f;
    float fw = (float)img_sz[n * 2 + 1] - 1.0f;
    float4 bx = reinterpret_cast<const float4*>(decBox)[(size_t)n * MM + m];
    float x1 = fminf(fmaxf(bx.x, 0.f), fw);
    float y1 = fminf(fmaxf(bx.y, 0.f), fh);
    float x2 = fminf(fmaxf(bx.z, 0.f), fw);
    float y2 = fminf(fmaxf(bx.w, 0.f), fh);
    keep = keep && ((x2 - x1 + 1.0f) >= 0.0f) && ((y2 - y1 + 1.0f) >= 0.0f); // MIN_SIZE=0
    float kf = keep ? 1.0f : 0.0f;

    if (m < TOPN) {
        float* ob = out;                         // boxes  [N,1000,4]
        float* os = out + (size_t)NN * TOPN * 4; // scores [N,1000]
        float* ol = os + (size_t)NN * TOPN;      // labels [N,1000]
        float* ok = ol + (size_t)NN * TOPN;      // keep   [N,1000]
        size_t idx = (size_t)n * TOPN + m;
        ob[idx * 4 + 0] = x1 * kf; ob[idx * 4 + 1] = y1 * kf;
        ob[idx * 4 + 2] = x2 * kf; ob[idx * 4 + 3] = y2 * kf;
        os[idx] = sc_m * kf;
        ol[idx] = (float)(clsA[n * MM + m] + 1);
        ok[idx] = kf;
    }
}

// ---------------- launch ------------------------------------------------------
extern "C" void kernel_launch(void* const* d_in, const int* in_sizes, int n_in,
                              void* d_out, int out_size, void* d_ws, size_t ws_size,
                              hipStream_t stream) {
    (void)in_sizes; (void)n_in; (void)out_size; (void)ws_size;
    char* ws = (char*)d_ws;
    size_t off = 0;
    auto take = [&](size_t bytes) -> void* {
        void* p = ws + off;
        off = (off + bytes + 255) & ~(size_t)255;
        return p;
    };
    float*     masked = (float*)take((size_t)NN * LL * 4);        // 77.8 MB
    int*       topIdx = (int*)take((size_t)NN * MM * 4);
    float*     topVal = (float*)take((size_t)NN * MM * 4);
    float*     decBox = (float*)take((size_t)NN * MM * 16);
    float*     offBox = (float*)take((size_t)NN * MM * 16);
    _Float16*  areaH  = (_Float16*)take((size_t)NN * MM * 2);
    float*     score  = (float*)take((size_t)NN * MM * 4);
    int*       clsA   = (int*)take((size_t)NN * MM * 4);
    unsigned*  validw = (unsigned*)take((size_t)NN * 32 * 4);
    unsigned*  supw   = (unsigned*)take((size_t)NN * 32 * 4);
    unsigned*  maskb  = (unsigned*)take((size_t)NN * MM * 32 * 4); // 2 MB

    const float* box_cls      = (const float*)d_in[0];
    const float* box_center   = (const float*)d_in[1];
    const float* border_cls   = (const float*)d_in[2];
    const float* border_delta = (const float*)d_in[3];
    const float* bd_based     = (const float*)d_in[4];
    const int*   img_sz       = (const int*)d_in[5];
    float* out = (float*)d_out;

    long total = (long)NN * KK * PP;
    k1_fuse<<<dim3((unsigned)((total + 255) / 256)), 256, 0, stream>>>(
        box_cls, box_center, border_cls, masked);
    k2_select<<<NN, 1024, 0, stream>>>(masked, topIdx, topVal);
    k3_decode<<<NN, 1024, 0, stream>>>(topIdx, topVal, border_delta, bd_based,
                                       decBox, offBox, areaH, score, clsA, validw, maskb);
    k4_iou<<<dim3(NN, 16), 256, 0, stream>>>(areaH, offBox, maskb);
    k5_nms<<<NN, 32, 0, stream>>>(validw, maskb, supw);
    k6_final<<<NN, 1024, 0, stream>>>(decBox, score, clsA, validw, supw, img_sz, out);
}
